// BiLSTMCRFTagger_86998857547983
// MI455X (gfx1250) — matrix operand order repaired and checked by reading, not compiled, not run on hardware
//
#include <hip/hip_runtime.h>
#include <hip/hip_bf16.h>

constexpr int kT      = 4096;
constexpr int kEmbed  = 256;
constexpr int kH2     = 256;
constexpr int kGates  = 1024;   // 4*H2
constexpr int kHidden = 512;
constexpr int kTag    = 51;
constexpr int kNTag   = 50;     // TAGSET-1 (CRF works on the first 50 tags)

typedef __attribute__((ext_vector_type(16))) __bf16 v16bf;
typedef __attribute__((ext_vector_type(8)))  float  v8f;

union FragBF {
  v16bf v;
  uint4 q[2];
};

__device__ __forceinline__ unsigned short f2bf(float f) {
  unsigned u = __float_as_uint(f);
  u += 0x7FFFu + ((u >> 16) & 1u);   // round-to-nearest-even
  return (unsigned short)(u >> 16);
}

__device__ __forceinline__ float sigmoidf_(float x) {
  return 1.0f / (1.0f + expf(-x));
}

// ---------------------------------------------------------------- prep: embedding gather -> bf16
__global__ void k_embed_bf16(const int* __restrict__ ids,
                             const float* __restrict__ table,
                             unsigned short* __restrict__ out) {
  int idx = blockIdx.x * blockDim.x + threadIdx.x;   // < kT*kEmbed
  int t = idx >> 8;
  int e = idx & 255;
  out[idx] = f2bf(table[(size_t)ids[t] * kEmbed + e]);
}

// ---------------------------------------------------------------- prep: f32 weights -> bf16
__global__ void k_f32_to_bf16(const float* __restrict__ in,
                              unsigned short* __restrict__ out, int n) {
  int i = blockIdx.x * blockDim.x + threadIdx.x;
  if (i < n) out[i] = f2bf(in[i]);
}

// ---------------------------------------------------------------- xg = embeds @ Wih.T + bih  (WMMA bf16->f32)
// block = 128 threads (4 waves). Block covers one 16-row M tile and 256 N cols
// (each wave owns 4 consecutive 16-col N tiles -> 32 WMMAs / wave, 4 indep acc chains).
// grid.x = T/16, grid.y = 2*kGates/256 (both directions fused on the N axis).
__global__ void __launch_bounds__(128) k_xg_wmma(
    const unsigned short* __restrict__ embBf,
    const unsigned short* __restrict__ WfBf,
    const unsigned short* __restrict__ WbBf,
    const float* __restrict__ bih_f,
    const float* __restrict__ bih_b,
    float* __restrict__ xg_f,
    float* __restrict__ xg_b) {
  __shared__ unsigned short As[16 * kEmbed];   // full 16x256 bf16 A panel (8 KB)

  const int tid   = threadIdx.x;
  const int lane  = tid & 31;
  const int wave  = tid >> 5;
  const int tileM = blockIdx.x;                        // 0..255
  const int gn    = blockIdx.y * 256 + wave * 64;      // 0..2047, 64-col span per wave
  const int dir   = (gn >= kGates) ? 1 : 0;
  const int nloc  = gn - dir * kGates;                 // base col within direction

  const unsigned short* W    = dir ? WbBf : WfBf;
  const float*          bias = dir ? bih_b : bih_f;
  float*                out  = dir ? xg_b : xg_f;

  // ---- stage the whole A panel once (coalesced uint4 = 8 bf16 per load) ----
#pragma unroll
  for (int i = 0; i < 4; ++i) {
    int e0 = (tid + i * 128) * 8;                      // < 16*256
    int r  = e0 >> 8;
    int c  = e0 & 255;
    *(uint4*)(&As[e0]) =
        *(const uint4*)(embBf + (size_t)(tileM * 16 + r) * kEmbed + c);
  }
  __syncthreads();

  // A-fragment addressing (contiguous k = 16h + s, same order as B; not the ISA table order):
  //   lane l: m = l&15 ; halves 0..15 -> K = aBase+0..15
  const int m     = lane & 15;
  const int aBase = (lane >> 4) * 16;
  // B-fragment addressing (16-bit B 32x16): lane l: n = l&15 ; half h -> K = bBase + h
  const int bBase = (lane >> 4) * 16;

  // Per-subtile B row pointers: B[k][n] = W[n][k]
  const unsigned short* wrow0 = W + (size_t)(nloc + 0 * 16 + (lane & 15)) * kEmbed;
  const unsigned short* wrow1 = W + (size_t)(nloc + 1 * 16 + (lane & 15)) * kEmbed;
  const unsigned short* wrow2 = W + (size_t)(nloc + 2 * 16 + (lane & 15)) * kEmbed;
  const unsigned short* wrow3 = W + (size_t)(nloc + 3 * 16 + (lane & 15)) * kEmbed;

  v8f acc0 = {0.f, 0.f, 0.f, 0.f, 0.f, 0.f, 0.f, 0.f};
  v8f acc1 = acc0, acc2 = acc0, acc3 = acc0;

#pragma unroll
  for (int kk = 0; kk < kEmbed; kk += 32) {
    FragBF a;
    a.q[0] = *(const uint4*)(&As[m * kEmbed + kk + aBase]);        // K aBase+0..7
    a.q[1] = *(const uint4*)(&As[m * kEmbed + kk + aBase + 8]);    // K aBase+8..15

    FragBF b0, b1, b2, b3;
    b0.q[0] = *(const uint4*)(wrow0 + kk + bBase);
    b0.q[1] = *(const uint4*)(wrow0 + kk + bBase + 8);
    b1.q[0] = *(const uint4*)(wrow1 + kk + bBase);
    b1.q[1] = *(const uint4*)(wrow1 + kk + bBase + 8);
    b2.q[0] = *(const uint4*)(wrow2 + kk + bBase);
    b2.q[1] = *(const uint4*)(wrow2 + kk + bBase + 8);
    b3.q[0] = *(const uint4*)(wrow3 + kk + bBase);
    b3.q[1] = *(const uint4*)(wrow3 + kk + bBase + 8);

    acc0 = __builtin_amdgcn_wmma_f32_16x16x32_bf16(false, a.v, false, b0.v,
                                                   (short)0, acc0, false, false);
    acc1 = __builtin_amdgcn_wmma_f32_16x16x32_bf16(false, a.v, false, b1.v,
                                                   (short)0, acc1, false, false);
    acc2 = __builtin_amdgcn_wmma_f32_16x16x32_bf16(false, a.v, false, b2.v,
                                                   (short)0, acc2, false, false);
    acc3 = __builtin_amdgcn_wmma_f32_16x16x32_bf16(false, a.v, false, b3.v,
                                                   (short)0, acc3, false, false);
  }

  // C/D layout: lane 0-15 -> M = r, lane 16-31 -> M = 8+r ; N = lane&15
  const int mHi = (lane >> 4) * 8;
#pragma unroll
  for (int j = 0; j < 4; ++j) {
    const v8f& acc = (j == 0) ? acc0 : (j == 1) ? acc1 : (j == 2) ? acc2 : acc3;
    const int  nc  = nloc + j * 16 + (lane & 15);
    const float bs = bias[nc];
#pragma unroll
    for (int r = 0; r < 8; ++r) {
      out[(size_t)(tileM * 16 + mHi + r) * kGates + nc] = acc[r] + bs;
    }
  }
}

// ---------------------------------------------------------------- LSTM recurrence (one block per direction)
__global__ void __launch_bounds__(1024) k_lstm(
    const float* __restrict__ xg_f, const float* __restrict__ xg_b,
    const float* __restrict__ Whh_f, const float* __restrict__ Whh_b,
    const float* __restrict__ bhh_f, const float* __restrict__ bhh_b,
    const float* __restrict__ h0, const float* __restrict__ c0,
    float* __restrict__ hidden) {
  __shared__ float hL[kH2];
  __shared__ float gL[kGates];

  const int tid = threadIdx.x;           // 0..1023 : one gate pre-activation each
  const int dir = blockIdx.x;            // 0 = forward, 1 = backward

  const float* xg  = dir ? xg_b : xg_f;
  const float* Whh = dir ? Whh_b : Whh_f;
  const float* bhh = dir ? bhh_b : bhh_f;

  float h = 0.f, c = 0.f;
  if (tid < kH2) {
    h = h0[dir * kH2 + tid];
    c = c0[dir * kH2 + tid];
    hL[tid] = h;
  }
  const float* wrow = Whh + (size_t)tid * kH2;
  const float  bb   = bhh[tid];
  __syncthreads();

  for (int s = 0; s < kT; ++s) {
    const int t = dir ? (kT - 1 - s) : s;
    if (s + 1 < kT) {   // hide the next step's xg fetch behind the matvec
      const int tn = dir ? (t - 1) : (t + 1);
      __builtin_prefetch(&xg[(size_t)tn * kGates + tid], 0, 1);
    }
    float sum = bb;
#pragma unroll 4
    for (int k = 0; k < kH2; k += 4) {
      float4 w = *(const float4*)(wrow + k);
      sum += w.x * hL[k] + w.y * hL[k + 1] + w.z * hL[k + 2] + w.w * hL[k + 3];
    }
    gL[tid] = xg[(size_t)t * kGates + tid] + sum;
    __syncthreads();
    if (tid < kH2) {
      const float gi = gL[tid];
      const float gf = gL[kH2 + tid];
      const float gg = gL[2 * kH2 + tid];
      const float go = gL[3 * kH2 + tid];
      c = sigmoidf_(gf) * c + sigmoidf_(gi) * tanhf(gg);
      h = sigmoidf_(go) * tanhf(c);
      hidden[(size_t)t * kHidden + dir * kH2 + tid] = h;
      hL[tid] = h;
    }
    __syncthreads();
  }
}

// ---------------------------------------------------------------- projection feats = hidden @ Wp.T + bp
__global__ void k_proj(const float* __restrict__ hidden,
                       const float* __restrict__ Wp,
                       const float* __restrict__ bp,
                       float* __restrict__ feats) {
  __shared__ float hr[kHidden];
  const int t = blockIdx.x;
  for (int k = threadIdx.x; k < kHidden; k += 64)
    hr[k] = hidden[(size_t)t * kHidden + k];
  __syncthreads();
  const int p = threadIdx.x;
  if (p < kTag) {
    const float* w = Wp + (size_t)p * kHidden;
    float s = bp[p];
    for (int k = 0; k < kHidden; k += 4) {
      float4 wv = *(const float4*)(w + k);
      s += wv.x * hr[k] + wv.y * hr[k + 1] + wv.z * hr[k + 2] + wv.w * hr[k + 3];
    }
    feats[(size_t)t * kTag + p] = s;
  }
}

// ---------------------------------------------------------------- Viterbi (1 block; backpointers as bytes in LDS)
__global__ void k_viterbi(const float* __restrict__ feats,
                          const float* __restrict__ trans,
                          float* __restrict__ out) {
  extern __shared__ unsigned char smem[];
  float* transL = (float*)smem;                        // kTag*kTag
  float* prevL  = transL + kTag * kTag;                // 64
  unsigned char* ptrL = (unsigned char*)(prevL + 64);  // (kT-1)*kNTag bytes

  const int tid = threadIdx.x;
  for (int i = tid; i < kTag * kTag; i += 64) transL[i] = trans[i];
  __syncthreads();
  // init_score = energy[0, -1, :] -> row index 49 of the :-1 slice
  if (tid < kNTag)
    prevL[tid] = transL[(kTag - 2) * kTag + tid] + feats[tid];
  __syncthreads();

  for (int t = 1; t < kT; ++t) {
    float best = -3.4e38f;
    int   arg  = 0;
    if (tid < kNTag) {
      for (int i = 0; i < kNTag; ++i) {
        float v = prevL[i] + transL[i * kTag + tid];
        if (v > best) { best = v; arg = i; }       // first-max tie-break like argmax
      }
      best += feats[(size_t)t * kTag + tid];
      ptrL[(size_t)(t - 1) * kNTag + tid] = (unsigned char)arg;
    }
    __syncthreads();
    if (tid < kNTag) prevL[tid] = best;
    __syncthreads();
  }

  if (tid == 0) {
    float bs = prevL[0];
    int   bj = 0;
    for (int j = 1; j < kNTag; ++j)
      if (prevL[j] > bs) { bs = prevL[j]; bj = j; }
    out[0] = bs;
    int* pred = (int*)out + 1;
    int  p = bj;
    pred[kT - 1] = p;
    for (int t = kT - 2; t >= 0; --t) {
      p = ptrL[(size_t)t * kNTag + p];
      pred[t] = p;
    }
  }
}

// ----------------------------------------------------------------
extern "C" void kernel_launch(void* const* d_in, const int* in_sizes, int n_in,
                              void* d_out, int out_size, void* d_ws, size_t ws_size,
                              hipStream_t stream) {
  const int*   sentence = (const int*)d_in[0];
  const float* table    = (const float*)d_in[1];
  const float* Wih_f    = (const float*)d_in[2];
  const float* Whh_f    = (const float*)d_in[3];
  const float* bih_f    = (const float*)d_in[4];
  const float* bhh_f    = (const float*)d_in[5];
  const float* Wih_b    = (const float*)d_in[6];
  const float* Whh_b    = (const float*)d_in[7];
  const float* bih_b    = (const float*)d_in[8];
  const float* bhh_b    = (const float*)d_in[9];
  const float* Wp       = (const float*)d_in[10];
  const float* bp       = (const float*)d_in[11];
  const float* trans    = (const float*)d_in[12];
  const float* h0       = (const float*)d_in[13];
  const float* c0       = (const float*)d_in[14];

  char* ws = (char*)d_ws;
  unsigned short* embBf = (unsigned short*)ws; ws += (size_t)kT * kEmbed * 2;
  unsigned short* WfBf  = (unsigned short*)ws; ws += (size_t)kGates * kEmbed * 2;
  unsigned short* WbBf  = (unsigned short*)ws; ws += (size_t)kGates * kEmbed * 2;
  float* xg_f   = (float*)ws; ws += (size_t)kT * kGates * 4;
  float* xg_b   = (float*)ws; ws += (size_t)kT * kGates * 4;
  float* hidden = (float*)ws; ws += (size_t)kT * kHidden * 4;
  float* feats  = (float*)ws; ws += (size_t)kT * kTag * 4;

  k_embed_bf16<<<(kT * kEmbed) / 256, 256, 0, stream>>>(sentence, table, embBf);
  k_f32_to_bf16<<<(kGates * kEmbed + 255) / 256, 256, 0, stream>>>(Wih_f, WfBf, kGates * kEmbed);
  k_f32_to_bf16<<<(kGates * kEmbed + 255) / 256, 256, 0, stream>>>(Wih_b, WbBf, kGates * kEmbed);

  dim3 g1(kT / 16, (2 * kGates) / 256);
  k_xg_wmma<<<g1, 128, 0, stream>>>(embBf, WfBf, WbBf, bih_f, bih_b, xg_f, xg_b);

  k_lstm<<<2, 1024, 0, stream>>>(xg_f, xg_b, Whh_f, Whh_b, bhh_f, bhh_b, h0, c0, hidden);

  k_proj<<<kT, 64, 0, stream>>>(hidden, Wp, bp, feats);

  size_t vitSmem = (size_t)kTag * kTag * 4 + 64 * 4 + (size_t)(kT - 1) * kNTag;
  k_viterbi<<<1, 64, vitSmem, stream>>>(feats, trans, (float*)d_out);
}
